// DeepSpeedSelfAttention_13958643712394
// MI455X (gfx1250) — matrix operand order repaired and checked by reading, not compiled, not run on hardware
//
#include <hip/hip_runtime.h>
#include <hip/hip_bf16.h>

// ---------------------------------------------------------------------------
// DeepSpeed-style self-attention block for MI455X (gfx1250, wave32, WMMA).
//   1) LayerNorm (fp32 -> bf16)
//   2) weight transpose+cast to [N][K] bf16 (one-time, L2-resident)
//   3) QKV GEMM via v_wmma_f32_16x16x32_bf16, fused head-split epilogue,
//      tiles staged with the Tensor Data Mover (TENSOR_LOAD_TO_LDS, padded
//      LDS rows via D# pad fields), double buffered
//   4) flash-attention per (b, head, 128-q-row tile),
//      K/V tiles staged with GLOBAL_LOAD_ASYNC_TO_LDS_B128
//   5) output projection GEMM (fp32 C + bias -> d_out)
// ---------------------------------------------------------------------------

typedef __attribute__((ext_vector_type(16))) __bf16 v16bf;
typedef __attribute__((ext_vector_type(8)))  __bf16 v8bf;
typedef __attribute__((ext_vector_type(8)))  float  v8f;
typedef __attribute__((ext_vector_type(4)))  unsigned int u32x4;
typedef __attribute__((ext_vector_type(8)))  int i32x8;
typedef __attribute__((ext_vector_type(4)))  int i32x4;

union Frag { v16bf v; v8bf h[2]; };

static __device__ __forceinline__ v8f wmma_bf16(const Frag& a, const Frag& b, v8f c) {
  // D = A(16x32 bf16) * B(32x16 bf16) + C(16x16 f32)
  return __builtin_amdgcn_wmma_f32_16x16x32_bf16(false, a.v, false, b.v,
                                                 (short)0, c, false, false);
}

static __device__ __forceinline__ v8f vzero8() {
  v8f z = {0.f, 0.f, 0.f, 0.f, 0.f, 0.f, 0.f, 0.f};
  return z;
}

// async 16-byte memory -> LDS copy (per-lane), tracked by ASYNCcnt.
static __device__ __forceinline__ void async_ld16(unsigned lds_off, const void* g) {
  asm volatile("global_load_async_to_lds_b128 %0, %1, off"
               : : "v"(lds_off), "v"(g) : "memory");
}
static __device__ __forceinline__ void wait_async0() {
  asm volatile("s_wait_asynccnt 0x0" ::: "memory");
}
static __device__ __forceinline__ unsigned lds_addr(const void* p) {
  return (unsigned)(size_t)p;   // flat LDS ptr: addr[31:0] = LDS offset
}

#if defined(__has_builtin)
#if __has_builtin(__builtin_amdgcn_tensor_load_to_lds)
#define HAVE_TDM 1
#endif
#endif

static __device__ __forceinline__ void wait_tensor0() {
#if defined(__has_builtin) && __has_builtin(__builtin_amdgcn_s_wait_tensorcnt)
  __builtin_amdgcn_s_wait_tensorcnt(0);
#else
  asm volatile("s_wait_tensorcnt 0x0" ::: "memory");
#endif
}

#ifdef HAVE_TDM
// Issue a 2D TDM load: tile_d1 rows x tile_d0 elems (2-byte elements) from a
// row-major tensor with row stride d0_stride (elems) into LDS at lds_byte_addr.
// LDS row padding: after every 2^(pad_interval+1) DWORDs insert (pad_amount+1)
// DWORDs of padding.
static __device__ __forceinline__ void tdm_load_2d(
    unsigned lds_byte_addr, const void* gptr,
    unsigned tensor_d0, unsigned tensor_d1, unsigned long long d0_stride,
    unsigned tile_d0, unsigned tile_d1,
    unsigned pad_interval, unsigned pad_amount) {
  unsigned long long ga = (unsigned long long)gptr;
  u32x4 g0;
  g0[0] = 1u;                                       // count=1, user D#
  g0[1] = lds_byte_addr;                            // lds_addr [63:32]
  g0[2] = (unsigned)(ga & 0xFFFFFFFFu);             // global_addr lo
  g0[3] = (unsigned)((ga >> 32) & 0x01FFFFFFu)      // global_addr[56:32]
        | (2u << 30);                               // type=2 ("image")
  i32x8 g1;
  g1[0] = (int)((1u << 16)                          // data_size = 2 bytes
              | (1u << 20)                          // pad_enable
              | (pad_interval << 22)
              | (pad_amount << 25));
  g1[1] = (int)((tensor_d0 & 0xFFFFu) << 16);       // tensor_dim0[15:0]
  g1[2] = (int)(((tensor_d0 >> 16) & 0xFFFFu)
              | ((tensor_d1 & 0xFFFFu) << 16));     // dim0 hi | dim1 lo
  g1[3] = (int)(((tensor_d1 >> 16) & 0xFFFFu)
              | (tile_d0 << 16));                   // dim1 hi | tile_dim0
  g1[4] = (int)tile_d1;                             // tile_dim1, tile_dim2=0
  g1[5] = (int)(unsigned)(d0_stride & 0xFFFFFFFFu); // dim0_stride lo
  g1[6] = (int)(unsigned)((d0_stride >> 32) & 0xFFFFu);
  g1[7] = 0;                                        // dim1_stride (unused, 2D)
  i32x4 z4 = {0, 0, 0, 0};
  i32x8 z8 = {0, 0, 0, 0, 0, 0, 0, 0};
  __builtin_amdgcn_tensor_load_to_lds(g0, g1, z4, z4, z8, 0);
}
#endif

// Problem constants
constexpr int Bsz = 2;
constexpr int Sq  = 2048;
constexpr int Hd  = 4096;
constexpr int NH  = 32;
constexpr int HD  = 128;          // head dim
constexpr int Mrows = Bsz * Sq;   // 4096
constexpr int N3 = 3 * Hd;        // 12288

// ---------------------------------------------------------------------------
// 1) LayerNorm: x[4096][4096] fp32 -> h bf16
// ---------------------------------------------------------------------------
__global__ void ln_kernel(const float* __restrict__ x,
                          const float* __restrict__ gw,
                          const float* __restrict__ gb,
                          __bf16* __restrict__ out) {
  const int row = blockIdx.x;
  const int t = threadIdx.x;
  const float* xr = x + (size_t)row * Hd;
  __shared__ float red[256];

  float s = 0.f;
  #pragma unroll
  for (int i = 0; i < Hd / 256; ++i) s += xr[t + i * 256];
  red[t] = s; __syncthreads();
  for (int off = 128; off > 0; off >>= 1) {
    if (t < off) red[t] += red[t + off];
    __syncthreads();
  }
  const float mu = red[0] * (1.f / Hd);
  __syncthreads();

  float v = 0.f;
  #pragma unroll
  for (int i = 0; i < Hd / 256; ++i) {
    float d = xr[t + i * 256] - mu;
    v += d * d;
  }
  red[t] = v; __syncthreads();
  for (int off = 128; off > 0; off >>= 1) {
    if (t < off) red[t] += red[t + off];
    __syncthreads();
  }
  const float rstd = rsqrtf(red[0] * (1.f / Hd) + 1e-5f);

  __bf16* orow = out + (size_t)row * Hd;
  #pragma unroll
  for (int i = 0; i < Hd / 256; ++i) {
    int c = t + i * 256;
    orow[c] = (__bf16)((xr[c] - mu) * rstd * gw[c] + gb[c]);
  }
}

// ---------------------------------------------------------------------------
// 2) transpose + cast: in fp32 [K][N] -> out bf16 [N][K]
// ---------------------------------------------------------------------------
__global__ void tcast_kernel(const float* __restrict__ in,
                             __bf16* __restrict__ out,
                             int K, int N) {
  size_t id = (size_t)blockIdx.x * blockDim.x + threadIdx.x;
  if (id >= (size_t)K * N) return;
  int n = (int)(id / K);
  int k = (int)(id % K);
  out[(size_t)n * K + k] = (__bf16)in[(size_t)k * N + n];
}

// ---------------------------------------------------------------------------
// combined additive bias per (b,h,kpos): (1-mask)*(-10000) + alibi[h][kpos]
// ---------------------------------------------------------------------------
__global__ void bias_kernel(const int* __restrict__ mask,
                            const float* __restrict__ alibi,
                            float* __restrict__ biasc) {
  int id = blockIdx.x * blockDim.x + threadIdx.x;   // [b][h][s]
  if (id >= Bsz * NH * Sq) return;
  int s = id & (Sq - 1);
  int h = (id / Sq) & (NH - 1);
  int b = id / (Sq * NH);
  biasc[id] = (1.f - (float)mask[b * Sq + s]) * (-10000.f) + alibi[h * Sq + s];
}

// ---------------------------------------------------------------------------
// GEMM: C[M][N] = A[M][K]bf16 * B^T[N][K]bf16 (+ bias)
// mode 0: C fp32 = acc + bias[n]   (out projection -> d_out)
// mode 1: fused QKV epilogue -> q/k (scaled, [bh][s][d]) and v^T ([bh][d][s])
// block: 256 thr = 8 waves; block tile 128(M) x 128(N); wave tile 64x32.
// Tiles staged by the Tensor Data Mover (wave 0 issues 2 descriptors/step),
// double buffered; D# pad fields give the 40-element padded LDS row stride
// (row 64B = 16 DW -> pad_interval=3; pad 4 DW -> pad_amount=3).
// ---------------------------------------------------------------------------
__global__ void gemm_kernel(const __bf16* __restrict__ A,
                            const __bf16* __restrict__ Bm,
                            const float* __restrict__ bias,
                            float* __restrict__ C,
                            int Mdim, int Ndim, int Kdim, int mode,
                            __bf16* __restrict__ qh,
                            __bf16* __restrict__ kh,
                            __bf16* __restrict__ vt,
                            float qscale) {
  const int n0 = blockIdx.x * 128;
  const int m0 = blockIdx.y * 128;
  const int t = threadIdx.x, L = t & 31, wv = t >> 5;
  const int wm = wv >> 2, wn = wv & 3;       // 2 x 4 wave grid

  __shared__ __align__(16) __bf16 ldsA[2][128 * 40];   // padded stride 40
  __shared__ __align__(16) __bf16 ldsB[2][128 * 40];

#ifdef HAVE_TDM
  auto stage = [&](int pb, int k0) {
    if (wv == 0) {
      tdm_load_2d(lds_addr(&ldsA[pb][0]), &A[(size_t)m0 * Kdim + k0],
                  (unsigned)Kdim, (unsigned)Mdim, (unsigned long long)Kdim,
                  32u, 128u, /*pad_interval=*/3u, /*pad_amount=*/3u);
      tdm_load_2d(lds_addr(&ldsB[pb][0]), &Bm[(size_t)n0 * Kdim + k0],
                  (unsigned)Kdim, (unsigned)Ndim, (unsigned long long)Kdim,
                  32u, 128u, 3u, 3u);
    }
  };
  auto wait_stage = [&]() { if (wv == 0) wait_tensor0(); };
#else
  const int c0row = (t + 0) >> 2,   c0col = ((t + 0) & 3) * 8;
  const int c1row = (t + 256) >> 2, c1col = ((t + 256) & 3) * 8;
  auto stage = [&](int pb, int k0) {
    async_ld16(lds_addr(&ldsA[pb][c0row * 40 + c0col]),
               &A[(size_t)(m0 + c0row) * Kdim + k0 + c0col]);
    async_ld16(lds_addr(&ldsB[pb][c0row * 40 + c0col]),
               &Bm[(size_t)(n0 + c0row) * Kdim + k0 + c0col]);
    async_ld16(lds_addr(&ldsA[pb][c1row * 40 + c1col]),
               &A[(size_t)(m0 + c1row) * Kdim + k0 + c1col]);
    async_ld16(lds_addr(&ldsB[pb][c1row * 40 + c1col]),
               &Bm[(size_t)(n0 + c1row) * Kdim + k0 + c1col]);
  };
  auto wait_stage = [&]() { wait_async0(); };
#endif

  v8f acc[4][2];
  #pragma unroll
  for (int fm = 0; fm < 4; ++fm)
    #pragma unroll
    for (int fn = 0; fn < 2; ++fn) acc[fm][fn] = vzero8();

  stage(0, 0);
  int pb = 0;
  for (int k0 = 0; k0 < Kdim; k0 += 32, pb ^= 1) {
    wait_stage();           // issuer's copies for buffer pb are in LDS
    __syncthreads();        // publish tile; prior reads of other buffer done
    if (k0 + 32 < Kdim) stage(pb ^ 1, k0 + 32);   // prefetch next tile

    Frag af[4], bf_[2];
    #pragma unroll
    for (int fm = 0; fm < 4; ++fm) {
      int row = wm * 64 + fm * 16 + (L & 15);
      int kb = (L >> 4) * 8;
      af[fm].h[0] = *(const v8bf*)&ldsA[pb][row * 40 + kb];
      af[fm].h[1] = *(const v8bf*)&ldsA[pb][row * 40 + kb + 16];
    }
    #pragma unroll
    for (int fn = 0; fn < 2; ++fn) {
      int n = wn * 32 + fn * 16 + (L & 15);
      int kg = (L >> 4) * 8;
      bf_[fn].h[0] = *(const v8bf*)&ldsB[pb][n * 40 + kg];
      bf_[fn].h[1] = *(const v8bf*)&ldsB[pb][n * 40 + kg + 16];
    }
    #pragma unroll
    for (int fm = 0; fm < 4; ++fm)
      #pragma unroll
      for (int fn = 0; fn < 2; ++fn)
        acc[fm][fn] = wmma_bf16(af[fm], bf_[fn], acc[fm][fn]);
  }

  // epilogue: C-fragment element (r,L): M = r + (L>>4)*8, N = L&15
  #pragma unroll
  for (int fm = 0; fm < 4; ++fm) {
    #pragma unroll
    for (int fn = 0; fn < 2; ++fn) {
      int nabs = n0 + wn * 32 + fn * 16 + (L & 15);
      float bv = bias[nabs];
      #pragma unroll
      for (int r = 0; r < 8; ++r) {
        int mabs = m0 + wm * 64 + fm * 16 + r + ((L >> 4) << 3);
        float val = acc[fm][fn][r] + bv;
        if (mode == 0) {
          C[(size_t)mabs * Ndim + nabs] = val;
        } else {
          int which = nabs >> 12;            // 0=q 1=k 2=v
          int within = nabs & 4095;
          int nh = within >> 7;
          int d  = within & 127;
          int b  = mabs >> 11;               // / Sq
          int s  = mabs & (Sq - 1);
          int head = b * NH + nh;
          if (which == 0)
            qh[((size_t)head * Sq + s) * HD + d] = (__bf16)(val * qscale);
          else if (which == 1)
            kh[((size_t)head * Sq + s) * HD + d] = (__bf16)(val * qscale);
          else
            vt[((size_t)head * HD + d) * Sq + s] = (__bf16)val;
        }
      }
    }
  }
}

// ---------------------------------------------------------------------------
// 4) flash attention: one block per (b*NH, 128 q rows). 8 waves x 16 q rows.
//    K / V^T tiles double-buffered in LDS via async memory->LDS copies.
// ---------------------------------------------------------------------------
__global__ void flash_kernel(const __bf16* __restrict__ qh,
                             const __bf16* __restrict__ kh,
                             const __bf16* __restrict__ vt,
                             const float* __restrict__ biasc,
                             __bf16* __restrict__ ctxb) {
  const int bh = blockIdx.x;                 // b*NH + h
  const int b = bh >> 5, h = bh & 31;
  const int q0 = blockIdx.y * 128;
  const int t = threadIdx.x, L = t & 31, wv = t >> 5;

  const __bf16* Q = qh + (size_t)bh * Sq * HD;
  const __bf16* K = kh + (size_t)bh * Sq * HD;
  const __bf16* V = vt + (size_t)bh * HD * Sq;
  const float* bias = biasc + (size_t)bh * Sq;

  __shared__ __align__(16) __bf16 ldsK[2][64 * 136];    // [kv][hd] pad 136
  __shared__ __align__(16) __bf16 ldsV[2][128 * 72];    // [hd][kv] pad 72
  __shared__ __align__(16) __bf16 ldsP[8 * 16 * 72];    // per-wave P region

  // stage one 64-kv tile of K and V^T (1024 chunks of 8 bf16 each)
  auto stageKV = [&](int pb, int kv0) {
    #pragma unroll
    for (int i = 0; i < 4; ++i) {
      int c = t + i * 256;
      int row = c >> 4, col = (c & 15) * 8;
      async_ld16(lds_addr(&ldsK[pb][row * 136 + col]),
                 &K[(size_t)(kv0 + row) * HD + col]);
    }
    #pragma unroll
    for (int i = 0; i < 4; ++i) {
      int c = t + i * 256;
      int row = c >> 3, col = (c & 7) * 8;
      async_ld16(lds_addr(&ldsV[pb][row * 72 + col]),
                 &V[(size_t)row * Sq + kv0 + col]);
    }
  };

  // Q fragments (A-operand), 16 rows x 128 hd, 4 K-steps of 32
  Frag qf[4];
  {
    int row = q0 + wv * 16 + (L & 15);
    int kb = (L >> 4) * 8;
    #pragma unroll
    for (int ks = 0; ks < 4; ++ks) {
      qf[ks].h[0] = *(const v8bf*)&Q[(size_t)row * HD + ks * 32 + kb];
      qf[ks].h[1] = *(const v8bf*)&Q[(size_t)row * HD + ks * 32 + kb + 16];
    }
  }

  v8f acc[8];
  #pragma unroll
  for (int f = 0; f < 8; ++f) acc[f] = vzero8();
  float mrow[8], lrow[8];
  #pragma unroll
  for (int r = 0; r < 8; ++r) { mrow[r] = -1e30f; lrow[r] = 0.f; }

  const int qbase = q0 + wv * 16 + ((L >> 4) << 3);
  const int kv_end = (q0 + 128 < Sq) ? (q0 + 128) : Sq;   // causal bound

  stageKV(0, 0);
  int pb = 0;
  for (int kv0 = 0; kv0 < kv_end; kv0 += 64, pb ^= 1) {
    wait_async0();
    __syncthreads();
    if (kv0 + 64 < kv_end) stageKV(pb ^ 1, kv0 + 64);

    // scores S = Q * K^T  (16 q rows x 64 kv per wave)
    v8f sf[4];
    #pragma unroll
    for (int fn = 0; fn < 4; ++fn) {
      v8f s = vzero8();
      int n = fn * 16 + (L & 15);
      #pragma unroll
      for (int ks = 0; ks < 4; ++ks) {
        Frag kb_;
        int kg = ks * 32 + (L >> 4) * 8;
        kb_.h[0] = *(const v8bf*)&ldsK[pb][n * 136 + kg];
        kb_.h[1] = *(const v8bf*)&ldsK[pb][n * 136 + kg + 16];
        s = wmma_bf16(qf[ks], kb_, s);
      }
      sf[fn] = s;
    }

    // bias + causal, online softmax (fp32)
    float mnew[8];
    #pragma unroll
    for (int r = 0; r < 8; ++r) mnew[r] = mrow[r];
    #pragma unroll
    for (int fn = 0; fn < 4; ++fn) {
      int kabs = kv0 + fn * 16 + (L & 15);
      float bsum = bias[kabs];
      #pragma unroll
      for (int r = 0; r < 8; ++r) {
        float val = sf[fn][r] + bsum + ((kabs > qbase + r) ? -10000.f : 0.f);
        sf[fn][r] = val;
        mnew[r] = fmaxf(mnew[r], val);
      }
    }
    #pragma unroll
    for (int r = 0; r < 8; ++r) {            // 16-lane row-max
      float v = mnew[r];
      v = fmaxf(v, __shfl_xor(v, 1, 32));
      v = fmaxf(v, __shfl_xor(v, 2, 32));
      v = fmaxf(v, __shfl_xor(v, 4, 32));
      v = fmaxf(v, __shfl_xor(v, 8, 32));
      mnew[r] = v;
    }
    float alpha[8], rsum[8];
    #pragma unroll
    for (int r = 0; r < 8; ++r) {
      alpha[r] = __expf(mrow[r] - mnew[r]);
      rsum[r] = 0.f;
      mrow[r] = mnew[r];
    }
    #pragma unroll
    for (int fn = 0; fn < 4; ++fn)
      #pragma unroll
      for (int r = 0; r < 8; ++r) {
        float p = __expf(sf[fn][r] - mnew[r]);
        sf[fn][r] = p;
        rsum[r] += p;
      }
    #pragma unroll
    for (int r = 0; r < 8; ++r) {            // 16-lane row-sum
      float v = rsum[r];
      v += __shfl_xor(v, 1, 32);
      v += __shfl_xor(v, 2, 32);
      v += __shfl_xor(v, 4, 32);
      v += __shfl_xor(v, 8, 32);
      lrow[r] = lrow[r] * alpha[r] + v;
    }
    #pragma unroll
    for (int f = 0; f < 8; ++f)
      #pragma unroll
      for (int r = 0; r < 8; ++r) acc[f][r] *= alpha[r];

    // P (C-layout) -> per-wave LDS -> A-fragment layout
    __bf16* P = &ldsP[wv * 16 * 72];
    #pragma unroll
    for (int fn = 0; fn < 4; ++fn) {
      int kl = fn * 16 + (L & 15);
      #pragma unroll
      for (int r = 0; r < 8; ++r) {
        int ql = r + ((L >> 4) << 3);
        P[ql * 72 + kl] = (__bf16)sf[fn][r];
      }
    }
    // ctx += P * V   (Kdim = 64 kv in 2 steps)
    #pragma unroll
    for (int kp = 0; kp < 2; ++kp) {
      Frag pf;
      int prow = (L & 15);
      int pkb = kp * 32 + (L >> 4) * 8;
      pf.h[0] = *(const v8bf*)&P[prow * 72 + pkb];
      pf.h[1] = *(const v8bf*)&P[prow * 72 + pkb + 16];
      #pragma unroll
      for (int fd = 0; fd < 8; ++fd) {
        Frag vf;
        int n = fd * 16 + (L & 15);
        int kg = kp * 32 + (L >> 4) * 8;
        vf.h[0] = *(const v8bf*)&ldsV[pb][n * 72 + kg];
        vf.h[1] = *(const v8bf*)&ldsV[pb][n * 72 + kg + 16];
        acc[fd] = wmma_bf16(pf, vf, acc[fd]);
      }
    }
  }

  // finalize: divide by l, write ctx bf16 interleaved [b*S+q][h*128+d]
  float inv[8];
  #pragma unroll
  for (int r = 0; r < 8; ++r) inv[r] = 1.f / lrow[r];
  #pragma unroll
  for (int fd = 0; fd < 8; ++fd) {
    int d = fd * 16 + (L & 15);
    #pragma unroll
    for (int r = 0; r < 8; ++r) {
      int qabs = qbase + r;
      ctxb[((size_t)(b * Sq + qabs)) * Hd + h * HD + d] =
          (__bf16)(acc[fd][r] * inv[r]);
    }
  }
}

// ---------------------------------------------------------------------------
// host launcher
// ---------------------------------------------------------------------------
extern "C" void kernel_launch(void* const* d_in, const int* in_sizes, int n_in,
                              void* d_out, int out_size, void* d_ws, size_t ws_size,
                              hipStream_t stream) {
  const float* x        = (const float*)d_in[0];
  const int*   in_mask  = (const int*)d_in[1];
  const float* alibi    = (const float*)d_in[2];
  const float* norm_w   = (const float*)d_in[3];
  const float* norm_b   = (const float*)d_in[4];
  const float* w_qkv    = (const float*)d_in[5];
  const float* b_qkv    = (const float*)d_in[6];
  const float* w_out    = (const float*)d_in[7];
  const float* b_out    = (const float*)d_in[8];
  float* out = (float*)d_out;

  // workspace carve-up (bytes)
  char* w = (char*)d_ws;
  size_t off = 0;
  auto alloc = [&](size_t bytes) { char* p = w + off; off = (off + bytes + 255) & ~(size_t)255; return p; };
  __bf16* h_b   = (__bf16*)alloc((size_t)Mrows * Hd * 2);         //  32 MB
  __bf16* wqkvT = (__bf16*)alloc((size_t)N3 * Hd * 2);            //  96 MB
  __bf16* woutT = (__bf16*)alloc((size_t)Hd * Hd * 2);            //  32 MB
  __bf16* qh    = (__bf16*)alloc((size_t)Bsz * NH * Sq * HD * 2); //  32 MB
  __bf16* kh    = (__bf16*)alloc((size_t)Bsz * NH * Sq * HD * 2); //  32 MB
  __bf16* vt    = (__bf16*)alloc((size_t)Bsz * NH * HD * Sq * 2); //  32 MB
  float*  biasc = (float*)alloc((size_t)Bsz * NH * Sq * 4);       // 0.5 MB
  __bf16* ctxb  = (__bf16*)alloc((size_t)Mrows * Hd * 2);         //  32 MB
  (void)ws_size;

  const float qscale = 1.0f / sqrtf(sqrtf((float)HD));

  // 1) layernorm
  ln_kernel<<<Mrows, 256, 0, stream>>>(x, norm_w, norm_b, h_b);

  // 2) weight transpose+cast
  {
    size_t tot = (size_t)Hd * N3;
    tcast_kernel<<<(int)((tot + 255) / 256), 256, 0, stream>>>(w_qkv, wqkvT, Hd, N3);
  }
  {
    size_t tot = (size_t)Hd * Hd;
    tcast_kernel<<<(int)((tot + 255) / 256), 256, 0, stream>>>(w_out, woutT, Hd, Hd);
  }

  // 3) QKV GEMM with fused head-split epilogue
  gemm_kernel<<<dim3(N3 / 128, Mrows / 128), 256, 0, stream>>>(
      h_b, wqkvT, b_qkv, nullptr, Mrows, N3, Hd, /*mode=*/1,
      qh, kh, vt, qscale);

  // combined mask/alibi bias
  bias_kernel<<<(Bsz * NH * Sq) / 256, 256, 0, stream>>>(in_mask, alibi, biasc);

  // 4) flash attention
  flash_kernel<<<dim3(Bsz * NH, Sq / 128), 256, 0, stream>>>(qh, kh, vt, biasc, ctxb);

  // 5) output projection -> d_out
  gemm_kernel<<<dim3(Hd / 128, Mrows / 128), 256, 0, stream>>>(
      ctxb, woutT, b_out, out, Mrows, Hd, Hd, /*mode=*/0,
      nullptr, nullptr, nullptr, 1.0f);
  (void)out_size; (void)n_in; (void)in_sizes;
}
